// PairwiseSimilarity_34454227648919
// MI455X (gfx1250) — matrix operand order repaired and checked by reading, not compiled, not run on hardware
//
#include <hip/hip_runtime.h>
#include <hip/hip_bf16.h>

// ---------------------------------------------------------------------------
// PairwiseSimilarity on MI455X (gfx1250):
//   C[q, o*162+t] = clamp( dot(normalize(Q[q]), normalize(R[o,t])), 0, 1 )
//   M=2048, N=10368, K=1024.
// Phase 1: fp32 row L2-normalize + convert to bf16 (workspace).
// Phase 2: LDS-tiled GEMM on v_wmma_f32_16x16x32_bf16 (fp32 accumulate).
//   Block tile 256x128, 8 waves, 64x64 per wave (4x4 WMMA tiles) so the
//   inner loop runs 16 WMMAs per 16 ds_load_b128 (1:1, LDS-bandwidth lean).
// ---------------------------------------------------------------------------

typedef __attribute__((ext_vector_type(16))) __bf16 v16bf;
typedef __attribute__((ext_vector_type(8)))  float  v8f;

#define MDIM 2048
#define NDIM 10368
#define KDIM 1024
#define BM   256
#define BN   128
#define BK   32
#define LDSTRIDE 40   // bf16 units per LDS row (32 data + 8 pad -> 80B, bank-friendly)

// float -> bf16, round-to-nearest-even
static __device__ __forceinline__ unsigned short f2bf(float f) {
    unsigned u = __float_as_uint(f);
    u += 0x7FFFu + ((u >> 16) & 1u);
    return (unsigned short)(u >> 16);
}

// One wave (32 lanes) per row of 1024 floats: sum of squares -> rsqrt -> scale -> bf16.
__global__ void __launch_bounds__(256)
normalize_rows_bf16_kernel(const float* __restrict__ src,
                           unsigned short* __restrict__ dst, int rows) {
    const int row  = blockIdx.x * 8 + (threadIdx.x >> 5);
    const int lane = threadIdx.x & 31;
    if (row >= rows) return;

    const float4* s = (const float4*)(src + (size_t)row * KDIM);
    float v0x[8], v0y[8], v0z[8], v0w[8];
    float ss = 0.0f;
#pragma unroll
    for (int c = 0; c < 8; ++c) {
        float4 v = s[lane + c * 32];
        v0x[c] = v.x; v0y[c] = v.y; v0z[c] = v.z; v0w[c] = v.w;
        ss += v.x * v.x + v.y * v.y + v.z * v.z + v.w * v.w;
    }
    // wave32 butterfly reduction
#pragma unroll
    for (int off = 16; off > 0; off >>= 1)
        ss += __shfl_xor(ss, off, 32);

    const float inv = 1.0f / fmaxf(sqrtf(ss), 1e-12f);   // matches x / max(||x||, eps)

    unsigned short* d = dst + (size_t)row * KDIM;
#pragma unroll
    for (int c = 0; c < 8; ++c) {
        ushort4 o;
        o.x = f2bf(v0x[c] * inv);
        o.y = f2bf(v0y[c] * inv);
        o.z = f2bf(v0z[c] * inv);
        o.w = f2bf(v0w[c] * inv);
        *(ushort4*)(d + (size_t)(lane + c * 32) * 4) = o;
    }
}

union FragU { uint4 u[2]; v16bf v; };

// 256 threads = 8 waves. Block tile 256(M)x128(N); wave grid 4(M)x2(N);
// wave tile 64x64 = 4x4 WMMA tiles -> 16 WMMAs per 16 ds_load_b128 per K-step.
__global__ void __launch_bounds__(256)
wmma_gemm_clamp_kernel(const unsigned short* __restrict__ Abf,   // [MDIM, KDIM] bf16 (normalized query)
                       const unsigned short* __restrict__ Bbf,   // [NDIM, KDIM] bf16 (normalized reference)
                       float* __restrict__ C) {                  // [MDIM, NDIM]
    __shared__ __align__(16) unsigned short lA[BM * LDSTRIDE];   // 20480 B
    __shared__ __align__(16) unsigned short lB[BN * LDSTRIDE];   // 10240 B

    const int tid  = threadIdx.x;
    const int wid  = tid >> 5;
    const int lane = tid & 31;
    const int g    = lane >> 4;     // K-half selector within the wave
    const int lm   = lane & 15;     // M row (A frag) / N col (B frag, C/D)

    const int m_base = blockIdx.y * BM;
    const int n_base = blockIdx.x * BN;
    const int wave_m = wid >> 1;    // 0..3 -> 64-row slabs
    const int wave_n = wid & 1;     // 0..1 -> 64-col slabs

    v8f acc[4][4];
#pragma unroll
    for (int i = 0; i < 4; ++i)
#pragma unroll
        for (int j = 0; j < 4; ++j) { v8f z = {}; acc[i][j] = z; }

    // global->LDS chunk maps (16B chunks of 8 bf16):
    //   A tile 256x32 -> 1024 chunks (4/thread); B tile 128x32 -> 512 chunks (2/thread)
    int arow[4], akc[4], brow[2], bkc[2];
#pragma unroll
    for (int cc = 0; cc < 4; ++cc) { int c = tid + cc * 256; arow[cc] = c >> 2; akc[cc] = c & 3; }
#pragma unroll
    for (int cc = 0; cc < 2; ++cc) { int c = tid + cc * 256; brow[cc] = c >> 2; bkc[cc] = c & 3; }

    for (int k0 = 0; k0 < KDIM; k0 += BK) {
        // stage tiles: issue global loads, fence previous compute, fill LDS
        uint4 ar[4], br[2];
#pragma unroll
        for (int cc = 0; cc < 4; ++cc)
            ar[cc] = *(const uint4*)(Abf + (size_t)(m_base + arow[cc]) * KDIM + k0 + akc[cc] * 8);
#pragma unroll
        for (int cc = 0; cc < 2; ++cc)
            br[cc] = *(const uint4*)(Bbf + (size_t)(n_base + brow[cc]) * KDIM + k0 + bkc[cc] * 8);
        __syncthreads();
#pragma unroll
        for (int cc = 0; cc < 4; ++cc)
            *(uint4*)(lA + arow[cc] * LDSTRIDE + akc[cc] * 8) = ar[cc];
#pragma unroll
        for (int cc = 0; cc < 2; ++cc)
            *(uint4*)(lB + brow[cc] * LDSTRIDE + bkc[cc] * 8) = br[cc];
        __syncthreads();

        // A fragments: lane(g,lm) holds row m=lm, K in [8g,8g+8) (V0-3) and [16+8g,..) (V4-7)
        FragU af[4];
#pragma unroll
        for (int i = 0; i < 4; ++i) {
            const unsigned short* rp = lA + (wave_m * 64 + i * 16 + lm) * LDSTRIDE;
            af[i].u[0] = *(const uint4*)(rp + 8 * g);
            af[i].u[1] = *(const uint4*)(rp + 16 + 8 * g);
        }
        // B fragments one at a time: lane(g,lm) holds col n=lm, K in [8g,8g+8) (V0-3) and [16+8g,..) (V4-7)
#pragma unroll
        for (int j = 0; j < 4; ++j) {
            FragU bfj;
            const unsigned short* rp = lB + (wave_n * 64 + j * 16 + lm) * LDSTRIDE;
            bfj.u[0] = *(const uint4*)(rp + 8 * g);
            bfj.u[1] = *(const uint4*)(rp + 16 + 8 * g);
#pragma unroll
            for (int i = 0; i < 4; ++i)
                acc[i][j] = __builtin_amdgcn_wmma_f32_16x16x32_bf16(
                    /*neg_a=*/false, af[i].v,
                    /*neg_b=*/false, bfj.v,
                    /*c_mod=*/(short)0, acc[i][j],
                    /*reuse_a=*/false, /*reuse_b=*/false);
        }
    }

    // epilogue: C/D layout -> VGPR v, lane(g,nl) holds C[8g+v][nl]; clamp to [0,1]
#pragma unroll
    for (int i = 0; i < 4; ++i) {
        const int mm = m_base + wave_m * 64 + i * 16 + g * 8;
#pragma unroll
        for (int j = 0; j < 4; ++j) {
            const int nn = n_base + wave_n * 64 + j * 16 + lm;
#pragma unroll
            for (int v = 0; v < 8; ++v) {
                float x = acc[i][j][v];
                x = fminf(fmaxf(x, 0.0f), 1.0f);
                C[(size_t)(mm + v) * NDIM + nn] = x;
            }
        }
    }
}

extern "C" void kernel_launch(void* const* d_in, const int* in_sizes, int n_in,
                              void* d_out, int out_size, void* d_ws, size_t ws_size,
                              hipStream_t stream) {
    const float* q = (const float*)d_in[0];   // [2048, 1024]
    const float* r = (const float*)d_in[1];   // [64, 162, 1024] -> rows n = o*162+t
    float* out = (float*)d_out;               // [2048, 10368]

    unsigned short* qbf = (unsigned short*)d_ws;                 // 4 MB
    unsigned short* rbf = qbf + (size_t)MDIM * KDIM;             // 20.25 MB more

    // Phase 1: normalize + bf16 convert (one wave per row, 8 rows per block)
    normalize_rows_bf16_kernel<<<MDIM / 8, 256, 0, stream>>>(q, qbf, MDIM);
    normalize_rows_bf16_kernel<<<NDIM / 8, 256, 0, stream>>>(r, rbf, NDIM);

    // Phase 2: WMMA GEMM + clamp. Grid: 81 N-tiles x 8 M-tiles.
    dim3 grid(NDIM / BN, MDIM / BM);
    wmma_gemm_clamp_kernel<<<grid, 256, 0, stream>>>(qbf, rbf, out);
}
